// RouteOptimizationModel_16681652978018
// MI455X (gfx1250) — matrix operand order repaired and checked by reading, not compiled, not run on hardware
//
#include <hip/hip_runtime.h>
#include <hip/hip_bf16.h>

typedef float     v2f  __attribute__((ext_vector_type(2)));
typedef float     v8f  __attribute__((ext_vector_type(8)));
typedef _Float16  v16h __attribute__((ext_vector_type(16)));

#define NN 8192
#define DH 128
#define NEG_SLOPE 0.2f

__device__ __forceinline__ float lrelu(float x){ return x > 0.f ? x : NEG_SLOPE * x; }

// gfx1250 async copy: global -> LDS, 16 bytes per lane, tracked by ASYNCcnt.
__device__ __forceinline__ void async_ld_b128(unsigned lds_addr, unsigned long long gaddr){
  asm volatile("global_load_async_to_lds_b128 %0, %1, off"
               :: "v"(lds_addr), "v"(gaddr) : "memory");
}
__device__ __forceinline__ void wait_asynccnt0(){
  asm volatile("s_wait_asynccnt 0x0" ::: "memory");
}

// ---------------- encoder stage 1: relu(nf @ W1 + b1), K=5 (VALU) -------------
__global__ void k_encoder1(const float* __restrict__ nf, const float* __restrict__ W1,
                           const float* __restrict__ b1, float* __restrict__ out){
  int idx = blockIdx.x * blockDim.x + threadIdx.x;   // NN*DH threads
  int i = idx >> 7, d = idx & 127;
  float s = b1[d];
  #pragma unroll
  for (int k = 0; k < 5; ++k) s += nf[i*5+k] * W1[k*DH + d];
  out[idx] = fmaxf(s, 0.f);
}

// ------------- fp32 WMMA GEMM: Y[NN,DH] = X[NN,DH] @ W[DH,DH] + bias ----------
// one wave per 16x16 output tile, K loop of 32 steps of v_wmma_f32_16x16x4_f32.
// TR variant additionally writes a transposed f16 copy YT16[col*NN + row]
// (contiguous-K B-tiles for the attention kernel).
template<bool TR>
__global__ __launch_bounds__(256) void k_gemm_wmma(const float* __restrict__ X,
    const float* __restrict__ W, const float* __restrict__ bias,
    float* __restrict__ Y, _Float16* __restrict__ YT16){
  int wid  = (blockIdx.x * 256 + threadIdx.x) >> 5;    // 0..4095
  int lane = threadIdx.x & 31;
  int tr = wid >> 3, tc = wid & 7;
  int m = lane & 15, hs = lane >> 4;
  int row = tr*16 + m;
  int col = tc*16 + m;
  v8f acc = {};
  #pragma unroll 4
  for (int kk = 0; kk < 32; ++kk){
    int kb = kk*4 + hs*2;
    v2f a = *(const v2f*)(X + row*DH + kb);             // A 16x4 f32 (2 VGPR)
    v2f b; b.x = W[(kb+0)*DH + col]; b.y = W[(kb+1)*DH + col];  // B 4x16 f32
    acc = __builtin_amdgcn_wmma_f32_16x16x4_f32(false, a, false, b,
                                                (short)0, acc, false, false);
  }
  #pragma unroll
  for (int r = 0; r < 8; ++r){
    int ro = tr*16 + r + hs*8;
    float v = acc[r] + bias[col];
    Y[ro*DH + col] = v;
    if (TR) YT16[(size_t)col * NN + ro] = (_Float16)v;
  }
}

// ---------------- s1 = xt@a1, s2 = xt@a2 (one wave per row) -------------------
__global__ void k_scores12(const float* __restrict__ xt, const float* __restrict__ a1,
                           const float* __restrict__ a2, float* __restrict__ s1,
                           float* __restrict__ s2){
  int wave = threadIdx.x >> 5, lane = threadIdx.x & 31;
  int row = blockIdx.x * (blockDim.x >> 5) + wave;
  float d1 = 0.f, d2 = 0.f;
  #pragma unroll
  for (int p = 0; p < 4; ++p){
    int d = lane + p*32;
    float x = xt[row*DH + d];
    d1 += x * a1[d]; d2 += x * a2[d];
  }
  for (int off = 16; off; off >>= 1){
    d1 += __shfl_xor(d1, off, 32);
    d2 += __shfl_xor(d2, off, 32);
  }
  if (lane == 0){ s1[row] = d1; s2[row] = d2; }
}

// ---------------- global top-2 (value+argmax) of s2 ---------------------------
__global__ void k_top2(const float* __restrict__ s2, float* __restrict__ out){
  __shared__ float v1s[256], v2s[256]; __shared__ int i1s[256];
  int t = threadIdx.x;
  float v1 = -3.0e38f, v2 = -3.0e38f; int i1 = -1;
  for (int p = 0; p < NN/256; ++p){
    int j = t + p*256;
    float v = s2[j];
    if (v > v1){ v2 = v1; v1 = v; i1 = j; }
    else if (v > v2){ v2 = v; }
  }
  v1s[t]=v1; v2s[t]=v2; i1s[t]=i1;
  __syncthreads();
  for (int s = 128; s; s >>= 1){
    if (t < s){
      float ov1=v1s[t+s], ov2=v2s[t+s]; int oi1=i1s[t+s];
      if (ov1 > v1s[t]){
        v2s[t] = fmaxf(v1s[t], ov2); v1s[t] = ov1; i1s[t] = oi1;
      } else {
        v2s[t] = fmaxf(v2s[t], ov1);
      }
    }
    __syncthreads();
  }
  if (t == 0){ out[0]=v1s[0]; out[1]=v2s[0]; out[2]=(float)i1s[0]; }
}

// ------- fused GAT attention: h_out = relu(xt + softmax(lrelu(c_i+t_j)) @ xt) -
// One wave per 16-row block (8 blocks-of-rows per workgroup). Row max is known
// analytically from the top-2 of s2 (leaky-relu is monotone) => single pass.
// B chunks (xt[j0:j0+32, :] in f16, 8 KB) are staged cooperatively into a
// double-buffered LDS tile with global_load_async_to_lds_b128 and shared by
// all 8 waves; accumulation via v_wmma_f32_16x16x32_f16.
__global__ __launch_bounds__(256) void k_gat_attn(const float* __restrict__ xt,
    const _Float16* __restrict__ xtT, const float* __restrict__ s1,
    const float* __restrict__ s2, const float* __restrict__ top2,
    const float* __restrict__ ab_p, float* __restrict__ hout){
  __shared__ float    lt[NN];            // all of s2: 32 KB
  __shared__ float    lz[8*16];
  __shared__ _Float16 lbuf[2*128*32];    // double-buffered B chunk: 2 x 8 KB
  for (int p = 0; p < NN/256; ++p) lt[threadIdx.x + p*256] = s2[threadIdx.x + p*256];

  int wave = threadIdx.x >> 5, lane = threadIdx.x & 31;
  int rb = blockIdx.x * 8 + wave;               // 16-row block
  int m = lane & 15, hs = lane >> 4;
  int i = rb * 16 + m;                          // this lane's row

  // this lane's slice of the cooperative async copy (1 KB per wave per chunk)
  int ccol = (wave << 4) + (lane >> 1);         // column 0..127
  int chalf = (lane & 1) << 4;                  // first/second 16 halfs of 32
  unsigned cdst = (unsigned)(unsigned long long)(&lbuf[ccol*32 + chalf]);
  const _Float16* csrc = xtT + (size_t)ccol * NN + chalf;

  // prologue: stage chunk j0=0 into buffer 0
  async_ld_b128(cdst,      (unsigned long long)csrc);
  async_ld_b128(cdst + 16, (unsigned long long)(csrc + 8));
  wait_asynccnt0();
  __syncthreads();

  float ab = ab_p[0];
  float c  = s1[i] + ab;
  float tmax = ((int)top2[2] == i) ? top2[1] : top2[0];   // exclude diagonal
  float mrow = lrelu(c + tmax);                 // exact row max (lrelu monotone)
  float zsum = 0.f;
  int kbA = hs * 16, kbB = hs * 16;

  v8f acc[8] = {};
  for (int it = 0; it < NN/32; ++it){
    int cur = it & 1;
    int j0 = it * 32;
    // prefetch next chunk into the other buffer (overlaps with WMMA below)
    if (it + 1 < NN/32){
      unsigned nd = cdst + (unsigned)((1-cur) ? (1)*(128*32*2) : 0) - (unsigned)(cur ? (128*32*2) : 0);
      // simpler: recompute destination for buffer (1-cur)
      nd = (unsigned)(unsigned long long)(&lbuf[(1-cur)*128*32 + ccol*32 + chalf]);
      const _Float16* ns = csrc + (it+1)*32;
      async_ld_b128(nd,      (unsigned long long)ns);
      async_ld_b128(nd + 16, (unsigned long long)(ns + 8));
    }
    const _Float16* bufc = &lbuf[cur*128*32];
    // A tile: 16x32 f16 weights, generated on the fly (16 exps per lane)
    v16h wv;
    #pragma unroll
    for (int v = 0; v < 8; ++v){
      int k0 = kbA + 2*v;
      #pragma unroll
      for (int e = 0; e < 2; ++e){
        int j = j0 + k0 + e;
        float sc = lrelu(c + lt[j]);
        float w  = __expf(sc - mrow);
        w = (j == i) ? 0.f : w;                 // diagonal masked out
        zsum += w;
        wv[2*v + e] = (_Float16)w;
      }
    }
    // 8 column tiles: B = xt[j0:j0+32, ct*16:+16] read from the LDS stage
    #pragma unroll
    for (int ct = 0; ct < 8; ++ct){
      int col = ct*16 + m;
      v16h bt = *(const v16h*)(bufc + col*32 + kbB);
      acc[ct] = __builtin_amdgcn_wmma_f32_16x16x32_f16(false, wv, false, bt,
                                                       (short)0, acc[ct], false, false);
    }
    wait_asynccnt0();      // next chunk landed
    __syncthreads();       // everyone done reading buf[cur], buf[1-cur] ready
  }
  float zrow = zsum + __shfl_xor(zsum, 16, 32);
  if (hs == 0) lz[wave*16 + m] = zrow;
  __syncthreads();
  #pragma unroll
  for (int r = 0; r < 8; ++r){
    int rowo = rb*16 + r + hs*8;
    float invz = 1.0f / lz[wave*16 + r + hs*8];
    #pragma unroll
    for (int ct = 0; ct < 8; ++ct){
      int col = ct*16 + m;
      float v = xt[rowo*DH + col] + acc[ct][r] * invz;
      hout[rowo*DH + col] = fmaxf(v, 0.f);
    }
  }
}

// ---------------- pointer head ------------------------------------------------
__global__ void k_q(const float* __restrict__ h, const float* __restrict__ Wq,
                    const float* __restrict__ bq, float* __restrict__ q){
  int d = threadIdx.x;
  float s = bq[d];
  for (int k = 0; k < DH; ++k) s += h[k] * Wq[k*DH + d];
  q[d] = s;
}
__global__ void k_uc(const float* __restrict__ q, const float* __restrict__ Wk,
                     const float* __restrict__ bk, float* __restrict__ uc){
  __shared__ float red[128];
  int t = threadIdx.x;
  float s = 0.f;
  for (int d = 0; d < DH; ++d) s += Wk[t*DH + d] * q[d];   // u = Wk @ q
  uc[t] = s;
  red[t] = q[t] * bk[t];
  __syncthreads();
  for (int st = 64; st; st >>= 1){ if (t < st) red[t] += red[t+st]; __syncthreads(); }
  if (t == 0) uc[DH] = red[0];                              // c = q . bk
}
__global__ void k_scores(const float* __restrict__ h, const float* __restrict__ uc,
                         float* __restrict__ out){
  int wave = threadIdx.x >> 5, lane = threadIdx.x & 31;
  int j = blockIdx.x * (blockDim.x >> 5) + wave;
  if (j >= NN - 1) return;
  const float* hr = h + (size_t)(1 + j) * DH;
  float s = 0.f;
  #pragma unroll
  for (int p = 0; p < 4; ++p) s += hr[lane + p*32] * uc[lane + p*32];
  for (int off = 16; off; off >>= 1) s += __shfl_xor(s, off, 32);
  if (lane == 0) out[j] = (s + uc[DH]) * 0.0883883476483184f;  // 1/sqrt(128)
}
__global__ void k_copyh(const float* __restrict__ h, float* __restrict__ out){
  int idx = blockIdx.x * blockDim.x + threadIdx.x;
  out[idx] = h[idx];   // b32 stores: d_out+8191 is only 4B-aligned
}

// ------------------------------------------------------------------------------
extern "C" void kernel_launch(void* const* d_in, const int* in_sizes, int n_in,
                              void* d_out, int out_size, void* d_ws, size_t ws_size,
                              hipStream_t stream){
  (void)in_sizes; (void)n_in; (void)out_size; (void)ws_size;
  const float* nf   = (const float*)d_in[0];
  // d_in[1] edge_index, d_in[2] edge_attr: unused by the reference math
  const float* Wne1 = (const float*)d_in[3];
  const float* bne1 = (const float*)d_in[4];
  const float* Wne2 = (const float*)d_in[5];
  const float* bne2 = (const float*)d_in[6];
  const float* gatW = (const float*)d_in[7];
  const float* gatb = (const float*)d_in[8];
  const float* ga1  = (const float*)d_in[9];
  const float* ga2  = (const float*)d_in[10];
  const float* gab  = (const float*)d_in[11];
  const float* Wq   = (const float*)d_in[12];
  const float* bq   = (const float*)d_in[13];
  const float* Wk   = (const float*)d_in[14];
  const float* bk   = (const float*)d_in[15];

  char* ws = (char*)d_ws;
  const size_t ND = (size_t)NN * DH;
  float*    hA  = (float*)(ws);
  float*    hB  = (float*)(ws + ND*4);
  float*    xt  = (float*)(ws + 2*ND*4);
  _Float16* xtT = (_Float16*)(ws + 3*ND*4);
  float*    s1  = (float*)(ws + 3*ND*4 + ND*2);
  float*    s2  = s1 + NN;
  float*    tp2 = s2 + NN;
  float*    qv  = tp2 + 16;
  float*    uc  = qv + 128;
  float* scores = (float*)d_out;
  float* outH   = scores + (NN - 1);

  // node encoder (xt buffer doubles as hmid scratch)
  k_encoder1<<<(NN*DH)/256, 256, 0, stream>>>(nf, Wne1, bne1, xt);
  k_gemm_wmma<false><<<512, 256, 0, stream>>>(xt, Wne2, bne2, hA, nullptr);

  float* hc = hA; float* hn = hB;
  for (int l = 0; l < 3; ++l){
    k_gemm_wmma<true><<<512, 256, 0, stream>>>(hc, gatW + (size_t)l*DH*DH, gatb + l*DH, xt, xtT);
    k_scores12<<<NN/8, 256, 0, stream>>>(xt, ga1 + l*DH, ga2 + l*DH, s1, s2);
    k_top2<<<1, 256, 0, stream>>>(s2, tp2);
    k_gat_attn<<<NN/128, 256, 0, stream>>>(xt, xtT, s1, s2, tp2, gab + l, hn);
    float* t = hc; hc = hn; hn = t;
  }

  k_q<<<1, 128, 0, stream>>>(hc, Wq, bq, qv);
  k_uc<<<1, 128, 0, stream>>>(qv, Wk, bk, uc);
  k_scores<<<2048, 128, 0, stream>>>(hc, uc, scores);
  k_copyh<<<(NN*DH)/256, 256, 0, stream>>>(hc, outH);
}
